// ResnetBlockMoE2D_85109071938199
// MI455X (gfx1250) — matrix operand order repaired and checked by reading, not compiled, not run on hardware
//
#include <hip/hip_runtime.h>
#include <hip/hip_bf16.h>

// ---------------------------------------------------------------------------
// ResNet MoE block for MI455X (gfx1250): wave32 + v_wmma_f32_16x16x32_bf16,
// async global->LDS gather for A tiles (global_load_async_to_lds_b128).
// ---------------------------------------------------------------------------

typedef __attribute__((ext_vector_type(16))) __bf16 bf16x16;
typedef __attribute__((ext_vector_type(8)))  float  f32x8;
typedef __attribute__((ext_vector_type(4)))  int    v4i;

typedef unsigned short u16;
typedef unsigned int   u32;

#define BATCH   4
#define CH      256     // C
#define SPAT    4096    // H*W = 64*64
#define TOKENS  (BATCH * SPAT)   // 16384
#define NEXP    8       // routed experts
#define IM      512     // moe intermediate
#define GROUPS  32
#define CPG     (CH / GROUPS)    // 8

// gfx1250 async global->LDS copy (ASYNCcnt path), if toolchain exposes it.
#if __has_builtin(__builtin_amdgcn_global_load_async_to_lds_b128)
#define HAVE_ASYNC_COPY 1
typedef __attribute__((address_space(1))) v4i* g4_ptr;   // global src
typedef __attribute__((address_space(3))) v4i* l4_ptr;   // LDS dst
#else
#define HAVE_ASYNC_COPY 0
#endif

// ---------------- helpers ----------------

__device__ __forceinline__ u16 f32_to_bf16(float f) {
  u32 u = __float_as_uint(f);
  u32 r = u + 0x7FFFu + ((u >> 16) & 1u);   // round-to-nearest-even
  return (u16)(r >> 16);
}
__device__ __forceinline__ float bf16_to_f32(u16 h) {
  return __uint_as_float(((u32)h) << 16);
}

union Frag16 { bf16x16 v; uint4 q[2]; };

__device__ __forceinline__ f32x8 zero8() {
  f32x8 z;
#pragma unroll
  for (int i = 0; i < 8; ++i) z[i] = 0.f;
  return z;
}

// A fragment, 16x32 bf16, row-major source (row stride ld halfs).
// Lanes 0-15: M=lane, K=kb+[0..8) and kb+[16..24); lanes 16-31: K=+8 shifted.
__device__ __forceinline__ bf16x16 load_a_frag(const u16* tile, int ld, int kb, int lane) {
  int m  = lane & 15;
  int ko = (lane >> 4) << 3;                 // 0 or 8
  const u16* p = tile + m * ld + kb + ko;
  Frag16 f;
  f.q[0] = *(const uint4*)(p);
  f.q[1] = *(const uint4*)(p + 16);
  return f.v;
}

// B fragment, 32x16 bf16, from transposed weight wt[n][k] (row stride ld halfs).
// Lanes 0-15: N=lane, K=kb+[0..8) and kb+[16..24); lanes 16-31: N=lane-16, K=+8 shifted.
__device__ __forceinline__ bf16x16 load_b_frag(const u16* wt, int ld, int nBase, int kb, int lane) {
  int n  = nBase + (lane & 15);
  int ks = kb + ((lane >> 4) << 3);          // 0 or 8
  const u16* p = wt + (size_t)n * ld + ks;
  Frag16 f;
  f.q[0] = *(const uint4*)(p);
  f.q[1] = *(const uint4*)(p + 16);
  return f.v;
}

__device__ __forceinline__ f32x8 wmma_bf16(bf16x16 a, bf16x16 b, f32x8 c) {
  return __builtin_amdgcn_wmma_f32_16x16x32_bf16(
      false, a, false, b, (short)0, c, false, false);
}

__device__ __forceinline__ float silu(float x) {
  return x * (1.f / (1.f + __expf(-x)));
}

// ---------------- kernel 0: zero expert counters ----------------

__global__ void init_kernel(int* __restrict__ cnt) {
  if (threadIdx.x < NEXP) cnt[threadIdx.x] = 0;
}

// ---------------- kernel 1: GroupNorm -> bf16 tokens [T, C] ----------------

__global__ __launch_bounds__(256)
void groupnorm_kernel(const float* __restrict__ x,
                      const float* __restrict__ scale,
                      const float* __restrict__ bias,
                      u16* __restrict__ t_bf16) {
  int b = blockIdx.x >> 5;
  int g = blockIdx.x & 31;
  int tid = threadIdx.x;
  const float* xb = x + ((size_t)b * CH + (size_t)g * CPG) * SPAT;

  float s = 0.f, ss = 0.f;
  for (int i = tid; i < CPG * SPAT; i += 256) {
    float v = xb[i];
    s += v; ss += v * v;
  }
  __shared__ float red0[256], red1[256];
  red0[tid] = s; red1[tid] = ss;
  __syncthreads();
  for (int off = 128; off > 0; off >>= 1) {
    if (tid < off) { red0[tid] += red0[tid + off]; red1[tid] += red1[tid + off]; }
    __syncthreads();
  }
  const float invN = 1.f / (float)(CPG * SPAT);
  float mu  = red0[0] * invN;
  float var = red1[0] * invN - mu * mu;
  float rs  = rsqrtf(var + 1e-6f);

  for (int i = tid; i < CPG * SPAT; i += 256) {
    int c  = g * CPG + (i >> 12);          // i / SPAT
    int sp = i & (SPAT - 1);
    float v = (xb[i] - mu) * rs * scale[c] + bias[c];
    int tok = b * SPAT + sp;
    t_bf16[(size_t)tok * CH + c] = f32_to_bf16(v);
  }
}

// ---------------- kernel 2: weight convert+transpose fp32[K,N] -> bf16[N,K] --

__global__ void convert_w_kernel(const float* __restrict__ src,
                                 u16* __restrict__ dst,
                                 int K, int N, int total) {
  int idx = blockIdx.x * 256 + threadIdx.x;
  if (idx >= total) return;
  int kn = K * N;
  int e  = idx / kn;
  int r  = idx - e * kn;
  int n  = r / K;                          // dst layout [e][n][k]
  int k  = r - n * K;
  dst[idx] = f32_to_bf16(src[(size_t)e * kn + (size_t)k * N + n]);
}

// ---------------- kernel 3: router (softmax, top-2, slot assignment) --------

__global__ __launch_bounds__(256)
void router_kernel(const u16* __restrict__ t_bf16,
                   const float* __restrict__ Wg,
                   int* __restrict__ cnt,
                   int* __restrict__ tokslot,
                   float* __restrict__ gate) {
  int t = blockIdx.x * 256 + threadIdx.x;
  if (t >= TOKENS) return;
  const u16* row = t_bf16 + (size_t)t * CH;

  float lg[NEXP];
#pragma unroll
  for (int e = 0; e < NEXP; ++e) lg[e] = 0.f;
  for (int c = 0; c < CH; ++c) {
    float tv = bf16_to_f32(row[c]);
    const float* wr = Wg + c * NEXP;
#pragma unroll
    for (int e = 0; e < NEXP; ++e) lg[e] += tv * wr[e];
  }
  float mx = lg[0];
#pragma unroll
  for (int e = 1; e < NEXP; ++e) mx = fmaxf(mx, lg[e]);
#pragma unroll
  for (int e = 0; e < NEXP; ++e) lg[e] = __expf(lg[e] - mx);
  // top-2 on unnormalized probs (softmax denom cancels in renormalization)
  int i0 = 0; float p0 = lg[0];
#pragma unroll
  for (int e = 1; e < NEXP; ++e) if (lg[e] > p0) { p0 = lg[e]; i0 = e; }
  int i1 = -1; float p1 = -1.f;
#pragma unroll
  for (int e = 0; e < NEXP; ++e) if (e != i0 && lg[e] > p1) { p1 = lg[e]; i1 = e; }
  float inv = 1.f / (p0 + p1);
  float w0 = p0 * inv, w1 = p1 * inv;

  int s0 = atomicAdd(&cnt[i0], 1);
  tokslot[i0 * TOKENS + s0] = t * 2 + 0;
  gate[i0 * TOKENS + s0]    = w0;
  int s1 = atomicAdd(&cnt[i1], 1);
  tokslot[i1 * TOKENS + s1] = t * 2 + 1;
  gate[i1 * TOKENS + s1]    = w1;
}

// ---------------- kernel 4: fused LlamaFFN via WMMA -------------------------
// 32-token tile, 8 waves. Stage 1: G=A@W1, U=A@W3 (each wave owns 64 cols of
// I=512, K-loop over C=256 in steps of 32 -> v_wmma_f32_16x16x32_bf16).
// SiLU-gate fused to bf16 H tile in LDS. Stage 2: Y = H@W2 (wave owns 32 of
// C=256 cols, K=512). ROUTED: A rows gathered via tokslot, output scaled by
// gate and scattered to ye[(2*token+k)*C]; else dense shared-expert path.
// A tile gathered into LDS through the gfx1250 async path when available.

template <bool ROUTED>
__global__ __launch_bounds__(256)
void ffn_kernel(const u16* __restrict__ t_bf16,
                const u16* __restrict__ wt1, const float* __restrict__ b1,
                const u16* __restrict__ wt3, const float* __restrict__ b3,
                const u16* __restrict__ wt2, const float* __restrict__ b2,
                const int* __restrict__ cnt,
                const int* __restrict__ tokslot,
                const float* __restrict__ gate,
                float* __restrict__ yout) {
  __shared__ __attribute__((aligned(16))) u16 As[32 * CH];   // 16 KB
  __shared__ __attribute__((aligned(16))) u16 Hs[32 * IM];   // 32 KB
  __shared__ int   toksS[32];
  __shared__ float gateS[32];

  const int tid  = threadIdx.x;
  const int lane = tid & 31;
  const int wave = tid >> 5;
  const int e    = ROUTED ? blockIdx.y : 0;
  const int tileBase = blockIdx.x * 32;

  const u16*   w1e = wt1 + (size_t)e * IM * CH;   // [n(I)][k(C)]
  const u16*   w3e = wt3 + (size_t)e * IM * CH;
  const u16*   w2e = wt2 + (size_t)e * CH * IM;   // [n(C)][k(I)]
  const float* b1e = b1 + e * IM;
  const float* b3e = b3 + e * IM;
  const float* b2e = b2 + e * CH;

  int nTok = 32;
  if (ROUTED) {
    nTok = cnt[e] - tileBase;
    if (nTok <= 0) return;                 // uniform per block
    if (nTok > 32) nTok = 32;
    if (tid < 32) {
      int slot  = tileBase + tid;
      int slotc = (tid < nTok) ? slot : tileBase;   // clamp pad rows
      toksS[tid] = tokslot[e * TOKENS + slotc];
      gateS[tid] = (tid < nTok) ? gate[e * TOKENS + slot] : 0.f;
    }
  } else {
    if (tid < 32) { toksS[tid] = (tileBase + tid) * 2; gateS[tid] = 1.f; }
  }

  // warm first B-strips of this block into cache while the A tile lands
  // (global_prefetch_b8; one line per thread across the wave's column strip)
  __builtin_prefetch(w1e + (size_t)(wave * 64 + (lane << 1)) * CH, 0, 1);
  __builtin_prefetch(w3e + (size_t)(wave * 64 + (lane << 1)) * CH, 0, 1);

  __syncthreads();

  // gather A tile: 32 rows x 256 bf16 (1024 x b128)
#if HAVE_ASYNC_COPY
  for (int i = tid; i < 32 * (CH / 8); i += 256) {
    int r  = i >> 5;                       // CH/8 == 32 uint4 per row
    int qo = i & 31;
    int token = toksS[r] >> 1;
    __builtin_amdgcn_global_load_async_to_lds_b128(
        (g4_ptr)(uintptr_t)&t_bf16[(size_t)token * CH + qo * 8],
        (l4_ptr)&As[r * CH + qo * 8], 0, 0);
  }
  asm volatile("s_wait_asynccnt 0x0" ::: "memory");
#else
  for (int i = tid; i < 32 * (CH / 8); i += 256) {
    int r  = i >> 5;
    int qo = i & 31;
    int token = toksS[r] >> 1;
    *(uint4*)&As[r * CH + qo * 8] =
        *(const uint4*)&t_bf16[(size_t)token * CH + qo * 8];
  }
#endif
  __syncthreads();

  // ---- stage 1 ----
  const int nBase0 = wave * 64;
  f32x8 accG[2][4], accU[2][4];
#pragma unroll
  for (int mi = 0; mi < 2; ++mi)
#pragma unroll
    for (int ni = 0; ni < 4; ++ni) { accG[mi][ni] = zero8(); accU[mi][ni] = zero8(); }

  for (int kb = 0; kb < CH; kb += 32) {
    bf16x16 a0 = load_a_frag(&As[0],        CH, kb, lane);   // rows 0..15
    bf16x16 a1 = load_a_frag(&As[16 * CH],  CH, kb, lane);   // rows 16..31
#pragma unroll
    for (int ni = 0; ni < 4; ++ni) {
      int n = nBase0 + ni * 16;
      bf16x16 bg = load_b_frag(w1e, CH, n, kb, lane);
      bf16x16 bu = load_b_frag(w3e, CH, n, kb, lane);
      accG[0][ni] = wmma_bf16(a0, bg, accG[0][ni]);
      accG[1][ni] = wmma_bf16(a1, bg, accG[1][ni]);
      accU[0][ni] = wmma_bf16(a0, bu, accU[0][ni]);
      accU[1][ni] = wmma_bf16(a1, bu, accU[1][ni]);
    }
  }

  // epilogue: H = silu(G + b1) * (U + b3), bf16 into LDS
#pragma unroll
  for (int mi = 0; mi < 2; ++mi) {
#pragma unroll
    for (int ni = 0; ni < 4; ++ni) {
      int ncol = nBase0 + ni * 16 + (lane & 15);
      float bg = b1e[ncol], bu = b3e[ncol];
      int rbase = mi * 16 + ((lane >> 4) << 3);   // +8 for hi lanes
#pragma unroll
      for (int r = 0; r < 8; ++r) {
        float g = accG[mi][ni][r] + bg;
        float u = accU[mi][ni][r] + bu;
        Hs[(rbase + r) * IM + ncol] = f32_to_bf16(silu(g) * u);
      }
    }
  }
  __syncthreads();

  // ---- stage 2 ----
  const int nBase2 = wave * 32;
  f32x8 accY[2][2];
#pragma unroll
  for (int mi = 0; mi < 2; ++mi)
#pragma unroll
    for (int nj = 0; nj < 2; ++nj) accY[mi][nj] = zero8();

  for (int kb = 0; kb < IM; kb += 32) {
    bf16x16 a0 = load_a_frag(&Hs[0],        IM, kb, lane);
    bf16x16 a1 = load_a_frag(&Hs[16 * IM],  IM, kb, lane);
#pragma unroll
    for (int nj = 0; nj < 2; ++nj) {
      bf16x16 bw = load_b_frag(w2e, IM, nBase2 + nj * 16, kb, lane);
      accY[0][nj] = wmma_bf16(a0, bw, accY[0][nj]);
      accY[1][nj] = wmma_bf16(a1, bw, accY[1][nj]);
    }
  }

  // epilogue: bias, gate scale, scatter
#pragma unroll
  for (int mi = 0; mi < 2; ++mi) {
#pragma unroll
    for (int nj = 0; nj < 2; ++nj) {
      int ncol = nBase2 + nj * 16 + (lane & 15);
      float bb = b2e[ncol];
      int rbase = mi * 16 + ((lane >> 4) << 3);
#pragma unroll
      for (int r = 0; r < 8; ++r) {
        int row = rbase + r;
        float v = accY[mi][nj][r] + bb;
        if (ROUTED) {
          if (row < nTok) {
            int tok2 = toksS[row];           // token*2 + k slot
            yout[(size_t)tok2 * CH + ncol] = v * gateS[row];
          }
        } else {
          int token = tileBase + row;
          yout[(size_t)token * CH + ncol] = v;
        }
      }
    }
  }
}

// ---------------- kernel 5: residual + shared + routed compose --------------

__global__ __launch_bounds__(256)
void compose_kernel(const float* __restrict__ x,
                    const float* __restrict__ shared_y,
                    const float* __restrict__ ye,
                    float* __restrict__ out) {
  size_t idx = (size_t)blockIdx.x * 256 + threadIdx.x;
  if (idx >= (size_t)BATCH * CH * SPAT) return;
  size_t sp = idx & (SPAT - 1);
  size_t c  = (idx >> 12) & (CH - 1);
  size_t b  = idx >> 20;
  size_t tok = b * SPAT + sp;
  out[idx] = x[idx]
           + shared_y[tok * CH + c]
           + ye[(tok * 2 + 0) * CH + c]
           + ye[(tok * 2 + 1) * CH + c];
}

// ---------------- launch ----------------------------------------------------

extern "C" void kernel_launch(void* const* d_in, const int* in_sizes, int n_in,
                              void* d_out, int out_size, void* d_ws, size_t ws_size,
                              hipStream_t stream) {
  (void)in_sizes; (void)n_in; (void)out_size; (void)ws_size;

  const float* x        = (const float*)d_in[0];
  const float* gn_scale = (const float*)d_in[1];
  const float* gn_bias  = (const float*)d_in[2];
  const float* Wg       = (const float*)d_in[3];
  const float* w1       = (const float*)d_in[4];
  const float* b1       = (const float*)d_in[5];
  const float* w3       = (const float*)d_in[6];
  const float* b3       = (const float*)d_in[7];
  const float* w2       = (const float*)d_in[8];
  const float* b2       = (const float*)d_in[9];
  const float* sw1      = (const float*)d_in[10];
  const float* sb1      = (const float*)d_in[11];
  const float* sw3      = (const float*)d_in[12];
  const float* sb3      = (const float*)d_in[13];
  const float* sw2      = (const float*)d_in[14];
  const float* sb2      = (const float*)d_in[15];
  float* out = (float*)d_out;

  // workspace carve-up (256B aligned slices)
  uintptr_t base = (uintptr_t)d_ws;
  auto carve = [&](size_t bytes) -> void* {
    void* p = (void*)base;
    base += (bytes + 255) & ~(size_t)255;
    return p;
  };
  u16*   t_bf16   = (u16*)  carve((size_t)TOKENS * CH * 2);          //  8 MB
  u16*   wt1      = (u16*)  carve((size_t)NEXP * IM * CH * 2);       //  2 MB
  u16*   wt3      = (u16*)  carve((size_t)NEXP * IM * CH * 2);       //  2 MB
  u16*   wt2      = (u16*)  carve((size_t)NEXP * CH * IM * 2);       //  2 MB
  u16*   swt1     = (u16*)  carve((size_t)IM * CH * 2);              // 256 KB
  u16*   swt3     = (u16*)  carve((size_t)IM * CH * 2);
  u16*   swt2     = (u16*)  carve((size_t)CH * IM * 2);
  int*   cnt      = (int*)  carve(256);
  int*   tokslot  = (int*)  carve((size_t)NEXP * TOKENS * 4);        // 512 KB
  float* gates    = (float*)carve((size_t)NEXP * TOKENS * 4);        // 512 KB
  float* shared_y = (float*)carve((size_t)TOKENS * CH * 4);          // 16 MB
  float* ye       = (float*)carve((size_t)2 * TOKENS * CH * 4);      // 32 MB

  init_kernel<<<1, 64, 0, stream>>>(cnt);

  groupnorm_kernel<<<BATCH * GROUPS, 256, 0, stream>>>(x, gn_scale, gn_bias, t_bf16);

  {
    int totE = NEXP * CH * IM;                           // 1,048,576
    int totS = CH * IM;                                  //   131,072
    convert_w_kernel<<<(totE + 255) / 256, 256, 0, stream>>>(w1,  wt1,  CH, IM, totE);
    convert_w_kernel<<<(totE + 255) / 256, 256, 0, stream>>>(w3,  wt3,  CH, IM, totE);
    convert_w_kernel<<<(totE + 255) / 256, 256, 0, stream>>>(w2,  wt2,  IM, CH, totE);
    convert_w_kernel<<<(totS + 255) / 256, 256, 0, stream>>>(sw1, swt1, CH, IM, totS);
    convert_w_kernel<<<(totS + 255) / 256, 256, 0, stream>>>(sw3, swt3, CH, IM, totS);
    convert_w_kernel<<<(totS + 255) / 256, 256, 0, stream>>>(sw2, swt2, IM, CH, totS);
  }

  router_kernel<<<TOKENS / 256, 256, 0, stream>>>(t_bf16, Wg, cnt, tokslot, gates);

  // shared expert: dense over all tokens
  ffn_kernel<false><<<TOKENS / 32, 256, 0, stream>>>(
      t_bf16, swt1, sb1, swt3, sb3, swt2, sb2, cnt, tokslot, gates, shared_y);

  // routed experts: gather-based grouped GEMM (over-launched, early exit)
  ffn_kernel<true><<<dim3(TOKENS / 32, NEXP), 256, 0, stream>>>(
      t_bf16, wt1, b1, wt3, b3, wt2, b2, cnt, tokslot, gates, ye);

  compose_kernel<<<(BATCH * CH * SPAT) / 256, 256, 0, stream>>>(x, shared_y, ye, out);
}
